// Block_78743930405073
// MI455X (gfx1250) — matrix operand order repaired and checked by reading, not compiled, not run on hardware
//
#include <hip/hip_runtime.h>

typedef __bf16 bf16_t;
typedef __attribute__((ext_vector_type(16))) __bf16 v16bf;
typedef __attribute__((ext_vector_type(8)))  float  v8f;
typedef __attribute__((ext_vector_type(4)))  unsigned uv4;
typedef __attribute__((ext_vector_type(8)))  unsigned uv8;

#define N_EMBD   1024
#define N_HEADS  16
#define HEAD_DIM 64
#define FF_DIM   4096
#define SEQ      2048
#define BATCH    4
#define ROWS     (BATCH * SEQ)   // 8192

union FragU {
    v16bf v;
    uint4 q[2];
};

// A fragment (16x32 bf16, MxK) from row-major LDS:
// lanes 0-15 hold M=0..15, K=0..15; lanes 16-31 K=16..31.
__device__ inline v16bf frag_a_rowmajor(const bf16_t* base, int ld) {
    const int lane = threadIdx.x & 31;
    const int m    = lane & 15;
    const int kg   = lane >> 4;
    FragU f;
    f.q[0] = *reinterpret_cast<const uint4*>(base + m * ld + kg * 16);
    f.q[1] = *reinterpret_cast<const uint4*>(base + m * ld + kg * 16 + 8);
    return f.v;
}

// B fragment (32x16 bf16, KxN) from LDS stored TRANSPOSED as [n][k], ldk = 32:
// lane holds column n = lane%16; lanes 0-15 K=0..15, lanes 16-31 K=16..31.
__device__ inline v16bf frag_b_transposed(const bf16_t* baseT) {
    const int lane = threadIdx.x & 31;
    const int n    = lane & 15;
    const int kh   = lane >> 4;
    const bf16_t* p = baseT + n * 32 + kh * 16;
    FragU f;
    f.q[0] = *reinterpret_cast<const uint4*>(p);
    f.q[1] = *reinterpret_cast<const uint4*>(p + 8);
    return f.v;
}

__device__ inline unsigned lds_byte_offset(const void* p) {
    // generic LDS address: addr[31:0] is the byte offset from the WG's LDS base
    return (unsigned)(unsigned long long)p;
}

// ---------------------------------------------------------------------------
// f32 -> bf16 weight conversion
// ---------------------------------------------------------------------------
__global__ void cvt_f32_bf16(const float* __restrict__ s, bf16_t* __restrict__ d, int n) {
    int i = blockIdx.x * blockDim.x + threadIdx.x;
    if (i < n) d[i] = (bf16_t)s[i];
}

// ---------------------------------------------------------------------------
// LayerNorm: one block per row of 1024, writes bf16
// ---------------------------------------------------------------------------
__global__ __launch_bounds__(256) void layernorm_kernel(
    const float* __restrict__ x, const float* __restrict__ g,
    const float* __restrict__ b, bf16_t* __restrict__ out)
{
    __shared__ float red[256];
    const int row = blockIdx.x;
    const float* xr = x + (long)row * N_EMBD;

    float s = 0.f;
    for (int i = threadIdx.x; i < N_EMBD; i += 256) s += xr[i];
    red[threadIdx.x] = s;
    __syncthreads();
    for (int o = 128; o > 0; o >>= 1) {
        if (threadIdx.x < o) red[threadIdx.x] += red[threadIdx.x + o];
        __syncthreads();
    }
    const float mu = red[0] * (1.f / N_EMBD);
    __syncthreads();

    float s2 = 0.f;
    for (int i = threadIdx.x; i < N_EMBD; i += 256) { float d = xr[i] - mu; s2 += d * d; }
    red[threadIdx.x] = s2;
    __syncthreads();
    for (int o = 128; o > 0; o >>= 1) {
        if (threadIdx.x < o) red[threadIdx.x] += red[threadIdx.x + o];
        __syncthreads();
    }
    const float rstd = rsqrtf(red[0] * (1.f / N_EMBD) + 1e-5f);

    bf16_t* orow = out + (long)row * N_EMBD;
    for (int i = threadIdx.x; i < N_EMBD; i += 256)
        orow[i] = (bf16_t)((xr[i] - mu) * rstd * g[i] + b[i]);
}

// ---------------------------------------------------------------------------
// bf16 matrix transpose: V[b][t][c] -> VT[b][c][t]  (LDS-tiled, padded)
// ---------------------------------------------------------------------------
__global__ __launch_bounds__(256) void transpose_bf16_kernel(
    const bf16_t* __restrict__ in, bf16_t* __restrict__ out)
{
    __shared__ bf16_t tile[32][33];
    const int b  = blockIdx.z;
    const int t0 = blockIdx.x * 32;
    const int c0 = blockIdx.y * 32;
    const int tx = threadIdx.x & 31;
    const int ty = threadIdx.x >> 5;   // 0..7
#pragma unroll
    for (int k = 0; k < 4; ++k) {
        const int t = ty + k * 8;
        tile[t][tx] = in[((size_t)b * SEQ + t0 + t) * N_EMBD + c0 + tx];
    }
    __syncthreads();
#pragma unroll
    for (int k = 0; k < 4; ++k) {
        const int c = ty + k * 8;
        out[((size_t)b * N_EMBD + c0 + c) * SEQ + t0 + tx] = tile[tx][c];
    }
}

// ---------------------------------------------------------------------------
// Tiled WMMA GEMM: C[M,N] = epilogue(A[M,K] @ W[K,N])
// block = 256 threads = 8 waves; block tile 128x64; wave tile 32x32; K-chunk 32.
// A tile (128x32 bf16) staged into LDS by the Tensor Data Mover: wave 0 issues
// one TENSOR_LOAD_TO_LDS with a 2D D# (data_size=2B, tile 32x128, stride K),
// waits TENSORcnt==0, and the workgroup barrier publishes the tile.
// EPI 0: store bf16(acc * scale)            (QKV; Q uses scale = 1/sqrt(D))
// EPI 1: store f32  (acc + bias + residual) (Wo / FF2 with residual add)
// EPI 2: store bf16(relu(acc + bias))       (FF1)
// ---------------------------------------------------------------------------
template <int EPI>
__global__ __launch_bounds__(256) void gemm_bf16_kernel(
    const bf16_t* __restrict__ A, const bf16_t* __restrict__ W,
    const float* __restrict__ bias, const float* __restrict__ residual,
    float* __restrict__ outF, bf16_t* __restrict__ outH,
    int M, int K, int N, float scale)
{
    __shared__ bf16_t sA[128 * 32];
    __shared__ bf16_t sB[64 * 32];   // transposed: [n][k]

    const int tid  = threadIdx.x;
    const int lane = tid & 31;
    const int wave = tid >> 5;
    const int wm   = wave >> 1;      // 0..3
    const int wn   = wave & 1;       // 0..1
    const long bm  = (long)blockIdx.x * 128;
    const long bn  = (long)blockIdx.y * 64;

    const unsigned lds_a = lds_byte_offset(&sA[0]);

    v8f acc[2][2];
#pragma unroll
    for (int mi = 0; mi < 2; ++mi)
#pragma unroll
        for (int ni = 0; ni < 2; ++ni)
#pragma unroll
            for (int e = 0; e < 8; ++e) acc[mi][ni][e] = 0.f;

    for (int k0 = 0; k0 < K; k0 += 32) {
        // --- stage A tile 128x32 via Tensor Data Mover (one DMA per K-step) ---
        if (wave == 0) {
            const unsigned long long ga =
                (unsigned long long)(const void*)(A + bm * (long)K + k0);
            uv4 g0;
            g0[0] = 1u;                                   // count=1 valid user D#
            g0[1] = lds_a;                                // lds_addr
            g0[2] = (unsigned)ga;                         // global_addr[31:0]
            g0[3] = (unsigned)((ga >> 32) & 0x01ffffffull) | (2u << 30); // addr[56:32] | type=2
            uv8 g1;
            g1[0] = 1u << 16;                             // data_size=1 (2 bytes)
            g1[1] = 32u << 16;                            // tensor_dim0 = 32
            g1[2] = 128u << 16;                           // tensor_dim1 = 128
            g1[3] = 32u << 16;                            // tile_dim0 = 32
            g1[4] = 128u;                                 // tile_dim1 = 128, tile_dim2 = 0
            g1[5] = (unsigned)K;                          // tensor_dim0_stride = K elems
            g1[6] = 0u;
            g1[7] = 0u;
            asm volatile("tensor_load_to_lds %0, %1" :: "s"(g0), "s"(g1) : "memory");
            __builtin_amdgcn_s_wait_tensorcnt(0);
        }
        // --- stage B tile 32x64, stored transposed [n][k] ---
        {
            const int k  = tid >> 3;           // 0..31
            const int n0 = (tid & 7) * 8;      // 0..56
            const uint4 wv = *reinterpret_cast<const uint4*>(
                W + (long)(k0 + k) * N + bn + n0);
            const bf16_t* wb = reinterpret_cast<const bf16_t*>(&wv);
#pragma unroll
            for (int j = 0; j < 8; ++j) sB[(n0 + j) * 32 + k] = wb[j];
        }
        __syncthreads();

        v16bf af[2], bfv[2];
#pragma unroll
        for (int mi = 0; mi < 2; ++mi)
            af[mi] = frag_a_rowmajor(sA + (wm * 32 + mi * 16) * 32, 32);
#pragma unroll
        for (int ni = 0; ni < 2; ++ni)
            bfv[ni] = frag_b_transposed(sB + (wn * 32 + ni * 16) * 32);

#pragma unroll
        for (int mi = 0; mi < 2; ++mi)
#pragma unroll
            for (int ni = 0; ni < 2; ++ni)
                acc[mi][ni] = __builtin_amdgcn_wmma_f32_16x16x32_bf16(
                    false, af[mi], false, bfv[ni], (short)0, acc[mi][ni], false, false);
        __syncthreads();
    }

    // --- epilogue ---
    const int colL = lane & 15;
    const int rowH = (lane >> 4) * 8;
#pragma unroll
    for (int mi = 0; mi < 2; ++mi) {
#pragma unroll
        for (int ni = 0; ni < 2; ++ni) {
            const long col = bn + wn * 32 + ni * 16 + colL;
#pragma unroll
            for (int r = 0; r < 8; ++r) {
                const long row = bm + wm * 32 + mi * 16 + rowH + r;
                const long idx = row * (long)N + col;
                const float v = acc[mi][ni][r] * scale;
                if (EPI == 0) {
                    outH[idx] = (bf16_t)v;
                } else if (EPI == 1) {
                    outF[idx] = v + bias[col] + residual[idx];
                } else {
                    const float t = v + bias[col];
                    outH[idx] = (bf16_t)(t > 0.f ? t : 0.f);
                }
            }
        }
    }
}

// ---------------------------------------------------------------------------
// Flash attention (causal). One wave per 16-query tile of one (batch, head).
// Q pre-scaled by 1/sqrt(D). Keys processed in chunks of 32 via 2 WMMA S-tiles,
// online softmax in f32 with shfl_xor row reductions, P staged through
// wave-private LDS (C-layout -> A-layout), O accumulated with WMMA.
// V consumed pre-transposed (VT[b][c][t]) so V fragments are contiguous loads.
// ---------------------------------------------------------------------------
__global__ __launch_bounds__(128) void attention_kernel(
    const bf16_t* __restrict__ Q, const bf16_t* __restrict__ Kt,
    const bf16_t* __restrict__ VT, bf16_t* __restrict__ Y)
{
    __shared__ bf16_t sP[4 * 16 * 32];
    const int lane = threadIdx.x & 31;
    const int wave = threadIdx.x >> 5;
    bf16_t* sPw = sP + wave * 512;

    const int gw    = blockIdx.x * 4 + wave;
    const int qt    = gw & (SEQ / 16 - 1);   // query tile 0..127
    const int hb    = gw >> 7;
    const int head  = hb & (N_HEADS - 1);
    const int batch = hb >> 4;
    const long rowbase = (long)batch * SEQ;
    const long hoff    = (long)head * HEAD_DIM;

    const int m   = lane & 15;
    const int kg  = lane >> 4;
    const int col = lane & 15;
    const int rb  = (lane >> 4) * 8;

    // Q fragments for d-chunks 0..31 and 32..63
    v16bf qf[2];
    {
        const long qrow = rowbase + qt * 16 + m;
#pragma unroll
        for (int dc = 0; dc < 2; ++dc) {
            const bf16_t* p = Q + qrow * N_EMBD + hoff + dc * 32;
            FragU f;
            f.q[0] = *reinterpret_cast<const uint4*>(p + kg * 16);
            f.q[1] = *reinterpret_cast<const uint4*>(p + kg * 16 + 8);
            qf[dc] = f.v;
        }
    }

    v8f o[4];
    float mrow[8], lrow[8];
#pragma unroll
    for (int ni = 0; ni < 4; ++ni)
#pragma unroll
        for (int e = 0; e < 8; ++e) o[ni][e] = 0.f;
#pragma unroll
    for (int r = 0; r < 8; ++r) { mrow[r] = -1e30f; lrow[r] = 0.f; }

    const int qend = (qt + 1) * 16;
    for (int s0 = 0; s0 < qend; s0 += 32) {
        // ---- S = Q · K^T for 32 keys (two 16x16 tiles) ----
        v8f sacc[2];
#pragma unroll
        for (int st = 0; st < 2; ++st) {
#pragma unroll
            for (int e = 0; e < 8; ++e) sacc[st][e] = 0.f;
            const long krow = rowbase + s0 + st * 16 + col;
#pragma unroll
            for (int dc = 0; dc < 2; ++dc) {
                const bf16_t* p = Kt + krow * N_EMBD + hoff + dc * 32 + kg * 16;
                FragU f;
                f.q[0] = *reinterpret_cast<const uint4*>(p);
                f.q[1] = *reinterpret_cast<const uint4*>(p + 8);
                sacc[st] = __builtin_amdgcn_wmma_f32_16x16x32_bf16(
                    false, qf[dc], false, f.v, (short)0, sacc[st], false, false);
            }
        }

        // ---- causal mask + online softmax; write P to LDS (row-major 16x32) ----
#pragma unroll
        for (int r = 0; r < 8; ++r) {
            const int t = qt * 16 + rb + r;
            float v0 = ((s0 + col)      <= t) ? sacc[0][r] : -1e30f;
            float v1 = ((s0 + 16 + col) <= t) ? sacc[1][r] : -1e30f;
            float mx = fmaxf(v0, v1);
#pragma unroll
            for (int sh = 8; sh >= 1; sh >>= 1) mx = fmaxf(mx, __shfl_xor(mx, sh, 32));
            const float mnew = fmaxf(mrow[r], mx);
            const float p0 = __expf(v0 - mnew);
            const float p1 = __expf(v1 - mnew);
            float rs = p0 + p1;
#pragma unroll
            for (int sh = 8; sh >= 1; sh >>= 1) rs += __shfl_xor(rs, sh, 32);
            const float resc = __expf(mrow[r] - mnew);
            lrow[r] = lrow[r] * resc + rs;
            mrow[r] = mnew;
#pragma unroll
            for (int ni = 0; ni < 4; ++ni) o[ni][r] *= resc;
            const int prow = rb + r;
            sPw[prow * 32 + col]      = (bf16_t)p0;
            sPw[prow * 32 + 16 + col] = (bf16_t)p1;
        }

        // ---- O += P · V (wave-private LDS round-trip; same-wave DS ops in order) ----
        const v16bf pf = frag_a_rowmajor(sPw, 32);
#pragma unroll
        for (int ni = 0; ni < 4; ++ni) {
            // B-layout: lane holds d-column hoff+ni*16+col; elems = 16 consecutive keys
            const bf16_t* vp = VT + ((size_t)batch * N_EMBD + hoff + ni * 16 + col) * SEQ
                                  + s0 + kg * 16;
            FragU f;
            f.q[0] = *reinterpret_cast<const uint4*>(vp);
            f.q[1] = *reinterpret_cast<const uint4*>(vp + 8);
            o[ni] = __builtin_amdgcn_wmma_f32_16x16x32_bf16(
                false, pf, false, f.v, (short)0, o[ni], false, false);
        }
    }

    // ---- normalize and store y (head-interleaved row-major, ready for Wo GEMM) ----
#pragma unroll
    for (int ni = 0; ni < 4; ++ni) {
#pragma unroll
        for (int r = 0; r < 8; ++r) {
            const long t = qt * 16 + rb + r;
            const float val = o[ni][r] / lrow[r];
            Y[(rowbase + t) * N_EMBD + hoff + ni * 16 + col] = (bf16_t)val;
        }
    }
}

// ---------------------------------------------------------------------------
// Host launcher
// ---------------------------------------------------------------------------
extern "C" void kernel_launch(void* const* d_in, const int* in_sizes, int n_in,
                              void* d_out, int out_size, void* d_ws, size_t ws_size,
                              hipStream_t stream)
{
    (void)in_sizes; (void)n_in; (void)out_size; (void)ws_size;
    const float* x   = (const float*)d_in[0];
    const float* Wq  = (const float*)d_in[1];
    const float* Wk  = (const float*)d_in[2];
    const float* Wv  = (const float*)d_in[3];
    const float* Wo  = (const float*)d_in[4];
    const float* bo  = (const float*)d_in[5];
    const float* W1  = (const float*)d_in[6];
    const float* b1  = (const float*)d_in[7];
    const float* W2  = (const float*)d_in[8];
    const float* b2  = (const float*)d_in[9];
    const float* g1  = (const float*)d_in[10];
    const float* be1 = (const float*)d_in[11];
    const float* g2  = (const float*)d_in[12];
    const float* be2 = (const float*)d_in[13];

    char* ws = (char*)d_ws;
    // workspace layout (bytes, 256-aligned); regions reused across phases
    bf16_t* wq_h = (bf16_t*)(ws + 0);           // 2 MiB
    bf16_t* wk_h = (bf16_t*)(ws + 2097152);
    bf16_t* wv_h = (bf16_t*)(ws + 4194304);
    bf16_t* wo_h = (bf16_t*)(ws + 6291456);
    bf16_t* w1_h = (bf16_t*)(ws + 8388608);     // 8 MiB
    bf16_t* w2_h = (bf16_t*)(ws + 16777216);    // 8 MiB
    float*  x1   = (float*) (ws + 25165824);    // 32 MiB
    bf16_t* h1   = (bf16_t*)(ws + 58720256);    // 16 MiB (reused as y)
    bf16_t* Qb   = (bf16_t*)(ws + 75497472);    // 16 MiB (reused as h2)
    bf16_t* Kb   = (bf16_t*)(ws + 92274688);    // 16 MiB
    bf16_t* Vb   = (bf16_t*)(ws + 109051904);   // 16 MiB
    bf16_t* VTb  = (bf16_t*)(ws + 125829120);   // 16 MiB (dead before FF1)
    bf16_t* yb   = h1;                          // y overwrites h1 (dead after QKV)
    bf16_t* h2   = Qb;                          // h2 overwrites Q (dead after attn)
    bf16_t* ub   = (bf16_t*)(ws + 92274688);    // 64 MiB over K/V/VT + tail

    // 1) weights -> bf16
    const int nw = N_EMBD * N_EMBD;      // 1,048,576
    const int nb = N_EMBD * FF_DIM;      // 4,194,304
    cvt_f32_bf16<<<(nw + 255) / 256, 256, 0, stream>>>(Wq, wq_h, nw);
    cvt_f32_bf16<<<(nw + 255) / 256, 256, 0, stream>>>(Wk, wk_h, nw);
    cvt_f32_bf16<<<(nw + 255) / 256, 256, 0, stream>>>(Wv, wv_h, nw);
    cvt_f32_bf16<<<(nw + 255) / 256, 256, 0, stream>>>(Wo, wo_h, nw);
    cvt_f32_bf16<<<(nb + 255) / 256, 256, 0, stream>>>(W1, w1_h, nb);
    cvt_f32_bf16<<<(nb + 255) / 256, 256, 0, stream>>>(W2, w2_h, nb);

    // 2) ln1
    layernorm_kernel<<<ROWS, 256, 0, stream>>>(x, g1, be1, h1);

    // 3) QKV (Q pre-scaled by 1/sqrt(HEAD_DIM) = 0.125)
    dim3 gcc(ROWS / 128, N_EMBD / 64);
    gemm_bf16_kernel<0><<<gcc, 256, 0, stream>>>(h1, wq_h, nullptr, nullptr, nullptr, Qb,
                                                 ROWS, N_EMBD, N_EMBD, 0.125f);
    gemm_bf16_kernel<0><<<gcc, 256, 0, stream>>>(h1, wk_h, nullptr, nullptr, nullptr, Kb,
                                                 ROWS, N_EMBD, N_EMBD, 1.0f);
    gemm_bf16_kernel<0><<<gcc, 256, 0, stream>>>(h1, wv_h, nullptr, nullptr, nullptr, Vb,
                                                 ROWS, N_EMBD, N_EMBD, 1.0f);

    // 3b) V -> VT (per-head column-major so attention V-fragments are contiguous)
    {
        dim3 gt(SEQ / 32, N_EMBD / 32, BATCH);
        transpose_bf16_kernel<<<gt, 256, 0, stream>>>(Vb, VTb);
    }

    // 4) causal flash attention -> y
    attention_kernel<<<(BATCH * N_HEADS * (SEQ / 16)) / 4, 128, 0, stream>>>(Qb, Kb, VTb, yb);

    // 5) x1 = x + y @ Wo + bo
    gemm_bf16_kernel<1><<<gcc, 256, 0, stream>>>(yb, wo_h, bo, x, x1, nullptr,
                                                 ROWS, N_EMBD, N_EMBD, 1.0f);

    // 6) ln2
    layernorm_kernel<<<ROWS, 256, 0, stream>>>(x1, g2, be2, h2);

    // 7) u = relu(h2 @ W1 + b1)
    dim3 gff(ROWS / 128, FF_DIM / 64);
    gemm_bf16_kernel<2><<<gff, 256, 0, stream>>>(h2, w1_h, b1, nullptr, nullptr, ub,
                                                 ROWS, N_EMBD, FF_DIM, 1.0f);

    // 8) out = x1 + u @ W2 + b2
    gemm_bf16_kernel<1><<<gcc, 256, 0, stream>>>(ub, w2_h, b2, x1, (float*)d_out, nullptr,
                                                 ROWS, FF_DIM, N_EMBD, 1.0f);
}
